// CMPNN_10625749090954
// MI455X (gfx1250) — matrix operand order repaired and checked by reading, not compiled, not run on hardware
//
#include <hip/hip_runtime.h>
#include <hip/hip_bf16.h>

// CMPNN fused pipeline for MI455X (gfx1250, wave32, WMMA).
// Heavy GEMMs use v_wmma_f32_16x16x32_bf16 with f32 accumulation.
// Weights pre-swizzled into per-lane WMMA B-fragment order (all LDS fragment
// reads are contiguous b128); W staged into LDS with async global->LDS DMA
// (ASYNCcnt) overlapped with the fused gather/subtract A staging.

typedef __attribute__((ext_vector_type(16))) __bf16        v16bf;
typedef __attribute__((ext_vector_type(2)))  __bf16        v2bf;
typedef __attribute__((ext_vector_type(8)))  float         v8f;
typedef __attribute__((ext_vector_type(8)))  unsigned int  v8u;
typedef __attribute__((ext_vector_type(4)))  unsigned int  v4u;
typedef __attribute__((ext_vector_type(2)))  unsigned int  v2u;

#define NND   30000      // nodes
#define NE    480000     // edges
#define D     128        // hidden
#define EDIM  16
#define OUTC  64

#define AROW  68         // A-tile LDS row stride in dwords (128 bf16 + pad, 16B aligned)

// ---- async global->LDS copy (GLOBAL_LOAD_ASYNC_TO_LDS_B128, ASYNCcnt) -----
__device__ __forceinline__ void async_copy_b128(void* lds_ptr, const void* gptr) {
    unsigned int lds_addr = (unsigned int)(unsigned long long)lds_ptr; // addr[31:0] = LDS offset
    unsigned long long gaddr = (unsigned long long)gptr;
    asm volatile("global_load_async_to_lds_b128 %0, %1, off"
                 :: "v"(lds_addr), "v"(gaddr)
                 : "memory");
}
__device__ __forceinline__ void wait_async0() {
    asm volatile("s_wait_asynccnt 0x0" ::: "memory");
}

// ---------------- weight conversion + B-fragment swizzle --------------------
// Output layout per 128x128 matrix: uint index u = ((kb*8 + ntile)*32 + lane)*8 + r
// holding the bf16 pair (W[k0][n], W[k0+1][n]) with
//   k0 = kb*32 + 8*(lane>>4) + 2*(r&3) + 16*(r>>2),  n = ntile*16 + (lane&15)
// Matrices: 0 = W_atom, 1..3 = W_seq[0..2].
__global__ void k_cvt_weights(const float* __restrict__ W_atom,
                              const float* __restrict__ W_seq,
                              unsigned int* __restrict__ Wswz) {
    int g = blockIdx.x * blockDim.x + threadIdx.x;   // 0..32767
    int mat = g >> 13;
    int u = g & 8191;
    int r = u & 7, lane = (u >> 3) & 31, nt = (u >> 8) & 7, kb = u >> 11;
    int hf = lane >> 4, l16 = lane & 15;
    int k0 = kb * 32 + 8 * hf + 2 * (r & 3) + 16 * (r >> 2);
    int n  = nt * 16 + l16;
    const float* Wsrc = (mat == 0) ? W_atom : (W_seq + (size_t)(mat - 1) * 16384);
    v2bf p = { (__bf16)Wsrc[k0 * D + n], (__bf16)Wsrc[(k0 + 1) * D + n] };
    Wswz[g] = __builtin_bit_cast(unsigned int, p);
}

// Per-lane fragment loaders -------------------------------------------------
__device__ __forceinline__ v16bf load_afrag(const unsigned int* AlU,
                                            int m, int kb, int hf) {
    const unsigned int* ap = AlU + m * AROW + kb * 16 + hf * 4;
    v4u lo = *(const v4u*)ap;          // k = kb*32 + [8*hf, 8*hf+8)
    v4u hi = *(const v4u*)(ap + 8);    // k = kb*32 + 16 + [8*hf, 8*hf+8)
    v8u au = { lo[0], lo[1], lo[2], lo[3], hi[0], hi[1], hi[2], hi[3] };
    return __builtin_bit_cast(v16bf, au);
}

__device__ __forceinline__ v16bf load_bfrag(const unsigned int* WlU,
                                            int kb, int nt, int lane) {
    const unsigned int* bp = WlU + (((kb * 8 + nt) * 32) + lane) * 8;
    v8u bu = *(const v8u*)bp;          // 32B contiguous per lane
    return __builtin_bit_cast(v16bf, bu);
}

// ---------------- node projection: x_proj = relu(x @ W_atom + b) ------------
__global__ __launch_bounds__(128)
void k_atom_proj(const float* __restrict__ x, const unsigned int* __restrict__ Wg,
                 const float* __restrict__ bias,
                 float* __restrict__ x_proj, float* __restrict__ h_atom) {
    __shared__ unsigned int Wl[8192];        // swizzled weights, 32 KB
    __shared__ unsigned int Al[16 * AROW];   // 16x128 bf16 A slab
    const int tid  = threadIdx.x;
    const int base = blockIdx.x * 16;        // 30000/16 = 1875 blocks

    #pragma unroll
    for (int i = 0; i < 16; ++i) {           // async W stage: 128 thr * 16 b128
        int off = (i * 128 + tid) * 4;
        async_copy_b128(Wl + off, Wg + off);
    }
    #pragma unroll
    for (int i = 0; i < 4; ++i) {            // stage A: fused load + bf16 pack
        int idx = i * 128 + tid;             // [0,512)
        int r = idx >> 5, c4 = (idx & 31) * 4;
        float4 va = *(const float4*)(x + (size_t)(base + r) * D + c4);
        v2bf p01 = { (__bf16)va.x, (__bf16)va.y };
        v2bf p23 = { (__bf16)va.z, (__bf16)va.w };
        v2u pp = { __builtin_bit_cast(unsigned int, p01),
                   __builtin_bit_cast(unsigned int, p23) };
        *(v2u*)(Al + r * AROW + (idx & 31) * 2) = pp;
    }
    wait_async0();
    __syncthreads();

    const int wave = tid >> 5, lane = tid & 31;
    const int hf = lane >> 4, l16 = lane & 15;
    const int nt0 = wave * 2;                // each wave: 32 output columns

    v8f acc0 = {}, acc1 = {};
    #pragma unroll
    for (int kb = 0; kb < 4; ++kb) {
        v16bf a  = load_afrag(Al, l16, kb, hf);
        v16bf b0 = load_bfrag(Wl, kb, nt0,     lane);
        v16bf b1 = load_bfrag(Wl, kb, nt0 + 1, lane);
        acc0 = __builtin_amdgcn_wmma_f32_16x16x32_bf16(false, a, false, b0,
                                                       (short)0, acc0, false, false);
        acc1 = __builtin_amdgcn_wmma_f32_16x16x32_bf16(false, a, false, b1,
                                                       (short)0, acc1, false, false);
    }
    const int n0 = nt0 * 16 + l16, n1 = n0 + 16;
    #pragma unroll
    for (int r = 0; r < 8; ++r) {            // C: VGPR r -> M = r + 8*hf
        int row = base + hf * 8 + r;
        float v0 = acc0[r] + bias[n0]; v0 = v0 > 0.f ? v0 : 0.f;
        float v1 = acc1[r] + bias[n1]; v1 = v1 > 0.f ? v1 : 0.f;
        x_proj[(size_t)row * D + n0] = v0; h_atom[(size_t)row * D + n0] = v0;
        x_proj[(size_t)row * D + n1] = v1; h_atom[(size_t)row * D + n1] = v1;
    }
}

// ---------------- bond init: h_bond = relu(edge_attr @ W_bond + b) ----------
__global__ void k_bond_init(const float* __restrict__ edge_attr,
                            const float* __restrict__ W_bond,
                            const float* __restrict__ b_bond,
                            float* __restrict__ h_bond) {
    int g = blockIdx.x * blockDim.x + threadIdx.x;   // NE*128 threads
    int e = g >> 7, d = g & 127;
    float s = b_bond[d];
    #pragma unroll
    for (int k = 0; k < EDIM; ++k) s += edge_attr[e * EDIM + k] * W_bond[k * D + d];
    h_bond[g] = s > 0.f ? s : 0.f;
}

// ---------------- aggregation -----------------------------------------------
__global__ void k_init_msg(float* __restrict__ msg_sum, int* __restrict__ msg_max) {
    int g = blockIdx.x * blockDim.x + threadIdx.x;   // NND*128
    msg_sum[g] = 0.f;
    msg_max[g] = 0;  // h_bond >= 0 (post-relu): int-bit max with init 0 == reference
}

__global__ void k_scatter(const float* __restrict__ h_bond,
                          const int* __restrict__ idx_i,
                          float* __restrict__ msg_sum, int* __restrict__ msg_max) {
    int g = blockIdx.x * blockDim.x + threadIdx.x;   // NE*128
    int e = g >> 7, d = g & 127;
    float v = h_bond[g];
    int node = idx_i[e];
    atomicAdd(&msg_sum[node * D + d], v);
    atomicMax(&msg_max[node * D + d], __float_as_int(v));  // valid: v >= 0
}

__global__ void k_node_update(float* __restrict__ h_atom,
                              const float* __restrict__ msg_sum,
                              const int* __restrict__ msg_max, int final_flag) {
    int g = blockIdx.x * blockDim.x + threadIdx.x;   // NND*128
    float msg = msg_sum[g] * __int_as_float(msg_max[g]);
    float h = h_atom[g];
    h_atom[g] = final_flag ? h * h * msg : h * msg;
}

__global__ void k_copy_head(const float* __restrict__ h_bond,
                            float* __restrict__ hb_head) {
    int g = blockIdx.x * blockDim.x + threadIdx.x;   // NND*128/4
    *(float4*)(hb_head + (size_t)g * 4) = *(const float4*)(h_bond + (size_t)g * 4);
}

// -------- fused edge update: h_bond = relu((h_atom[i] - hb_head[j]) @ W + b)
__global__ __launch_bounds__(256)
void k_edge_gemm(const float* __restrict__ h_atom, const float* __restrict__ hb_head,
                 const int* __restrict__ idx_i, const int* __restrict__ idx_j,
                 const unsigned int* __restrict__ Wg, const float* __restrict__ bias,
                 float* __restrict__ h_bond) {
    __shared__ unsigned int Wl[8192];        // swizzled weights, 32 KB
    __shared__ unsigned int Al[32 * AROW];   // 32x128 bf16 gathered A slab
    __shared__ int si[32], sj[32];
    const int tid  = threadIdx.x;
    const int base = blockIdx.x * 32;        // 480000/32 = 15000 blocks

    #pragma unroll
    for (int i = 0; i < 8; ++i) {            // async W stage: 256 thr * 8 b128
        int off = (i * 256 + tid) * 4;
        async_copy_b128(Wl + off, Wg + off);
    }
    if (tid < 32) { si[tid] = idx_i[base + tid]; sj[tid] = idx_j[base + tid]; }
    __syncthreads();                          // si/sj visible
    #pragma unroll
    for (int i = 0; i < 4; ++i) {            // fused gather + subtract + bf16 pack
        int idx = i * 256 + tid;             // [0,1024)
        int r = idx >> 5, c4 = (idx & 31) * 4;
        float4 va = *(const float4*)(h_atom  + (size_t)si[r] * D + c4);
        float4 vb = *(const float4*)(hb_head + (size_t)sj[r] * D + c4);
        v2bf p01 = { (__bf16)(va.x - vb.x), (__bf16)(va.y - vb.y) };
        v2bf p23 = { (__bf16)(va.z - vb.z), (__bf16)(va.w - vb.w) };
        v2u pp = { __builtin_bit_cast(unsigned int, p01),
                   __builtin_bit_cast(unsigned int, p23) };
        *(v2u*)(Al + r * AROW + (idx & 31) * 2) = pp;
    }
    wait_async0();
    __syncthreads();

    const int wave = tid >> 5, lane = tid & 31;
    const int hf = lane >> 4, l16 = lane & 15;
    const int msub = wave & 1;               // 2 row-subtiles of 16 edges
    const int nt0 = (wave >> 1) * 2;         // 4 column slabs of 32
    const int m = msub * 16 + l16;

    v8f acc0 = {}, acc1 = {};
    #pragma unroll
    for (int kb = 0; kb < 4; ++kb) {
        v16bf a  = load_afrag(Al, m, kb, hf);
        v16bf b0 = load_bfrag(Wl, kb, nt0,     lane);
        v16bf b1 = load_bfrag(Wl, kb, nt0 + 1, lane);
        acc0 = __builtin_amdgcn_wmma_f32_16x16x32_bf16(false, a, false, b0,
                                                       (short)0, acc0, false, false);
        acc1 = __builtin_amdgcn_wmma_f32_16x16x32_bf16(false, a, false, b1,
                                                       (short)0, acc1, false, false);
    }
    const int n0 = nt0 * 16 + l16, n1 = n0 + 16;
    #pragma unroll
    for (int r = 0; r < 8; ++r) {
        int row = base + msub * 16 + hf * 8 + r;
        float v0 = acc0[r] + bias[n0]; v0 = v0 > 0.f ? v0 : 0.f;
        float v1 = acc1[r] + bias[n1]; v1 = v1 > 0.f ? v1 : 0.f;
        h_bond[(size_t)row * D + n0] = v0;
        h_bond[(size_t)row * D + n1] = v1;
    }
}

// ---------------- final projection: out = [h_atom, x_proj] @ W_lin + b ------
__global__ void k_final(const float* __restrict__ h_atom,
                        const float* __restrict__ x_proj,
                        const float* __restrict__ W_lin,
                        const float* __restrict__ b_lin,
                        float* __restrict__ out) {
    int g = blockIdx.x * blockDim.x + threadIdx.x;   // NND*64
    int n = g >> 6, o = g & 63;
    float s = b_lin[o];
    for (int k = 0; k < D; ++k) s += h_atom[(size_t)n * D + k] * W_lin[k * OUTC + o];
    for (int k = 0; k < D; ++k) s += x_proj[(size_t)n * D + k] * W_lin[(D + k) * OUTC + o];
    out[g] = s;
}

extern "C" void kernel_launch(void* const* d_in, const int* in_sizes, int n_in,
                              void* d_out, int out_size, void* d_ws, size_t ws_size,
                              hipStream_t stream) {
    const float* x         = (const float*)d_in[0];
    const int*   edge_idx  = (const int*)d_in[1];
    const float* edge_attr = (const float*)d_in[2];
    const float* W_atom    = (const float*)d_in[3];
    const float* b_atom    = (const float*)d_in[4];
    const float* W_bond    = (const float*)d_in[5];
    const float* b_bond    = (const float*)d_in[6];
    const float* W_seq     = (const float*)d_in[7];
    const float* b_seq     = (const float*)d_in[8];
    const float* W_lin     = (const float*)d_in[9];
    const float* b_lin     = (const float*)d_in[10];
    const int* idx_i = edge_idx;
    const int* idx_j = edge_idx + NE;

    // workspace carve-up
    char* ws = (char*)d_ws;
    size_t off = 0;
    float* x_proj  = (float*)(ws + off); off += (size_t)NND * D * 4;
    float* h_atom  = (float*)(ws + off); off += (size_t)NND * D * 4;
    float* h_bond  = (float*)(ws + off); off += (size_t)NE  * D * 4;
    float* hb_head = (float*)(ws + off); off += (size_t)NND * D * 4;
    float* msg_sum = (float*)(ws + off); off += (size_t)NND * D * 4;
    int*   msg_max = (int*)  (ws + off); off += (size_t)NND * D * 4;
    unsigned int* Wswz = (unsigned int*)(ws + off); off += (size_t)32768 * 4;

    const int TB = 256;
    const int gridNE = NE * D / TB;    // 240000
    const int gridNN = NND * D / TB;   // 15000

    k_cvt_weights<<<32768 / TB, TB, 0, stream>>>(W_atom, W_seq, Wswz);
    k_atom_proj<<<NND / 16, 128, 0, stream>>>(x, Wswz, b_atom, x_proj, h_atom);
    k_bond_init<<<gridNE, TB, 0, stream>>>(edge_attr, W_bond, b_bond, h_bond);

    for (int l = 0; l < 3; ++l) {
        k_init_msg<<<gridNN, TB, 0, stream>>>(msg_sum, msg_max);
        k_scatter<<<gridNE, TB, 0, stream>>>(h_bond, idx_i, msg_sum, msg_max);
        k_node_update<<<gridNN, TB, 0, stream>>>(h_atom, msg_sum, msg_max, 0);
        k_copy_head<<<gridNN / 4, TB, 0, stream>>>(h_bond, hb_head);
        k_edge_gemm<<<NE / 32, 256, 0, stream>>>(h_atom, hb_head, idx_i, idx_j,
                                                 Wswz + (size_t)8192 * (1 + l),
                                                 b_seq + (size_t)l * D, h_bond);
    }
    // final aggregation: h_atom = h_atom^2 * msg
    k_init_msg<<<gridNN, TB, 0, stream>>>(msg_sum, msg_max);
    k_scatter<<<gridNE, TB, 0, stream>>>(h_bond, idx_i, msg_sum, msg_max);
    k_node_update<<<gridNN, TB, 0, stream>>>(h_atom, msg_sum, msg_max, 1);

    k_final<<<NND * OUTC / TB, TB, 0, stream>>>(h_atom, x_proj, W_lin, b_lin,
                                                (float*)d_out);
}
